// SimpleVectorQuantizer_7876970021322
// MI455X (gfx1250) — compile-verified, matrix operand order checked by reading, not run on hardware
//
#include <hip/hip_runtime.h>

#define B_    8
#define N_    2048
#define D_    256
#define K_    8192
#define BN_   (B_ * N_)
#define INV_T 100.0f   /* 1 / 0.01 entropy temperature */
#define BETA_ 0.25f
#define ENTW_ 0.1f

typedef __attribute__((ext_vector_type(8)))  float  v8f;
typedef __attribute__((ext_vector_type(8)))  __bf16 v8bf;
typedef __attribute__((ext_vector_type(16))) __bf16 v16bf;

// ---------------------------------------------------------------------------
// Row squared-norms: one 256-thread block per row of a [nrows, 256] matrix.
// (exact fp32 norms — only the dot product is approximated)
// ---------------------------------------------------------------------------
__global__ __launch_bounds__(256) void vq_rownorm(const float* __restrict__ x,
                                                  float* __restrict__ out) {
  __shared__ float red[256];
  const int row = blockIdx.x;
  float v = x[(size_t)row * D_ + threadIdx.x];
  red[threadIdx.x] = v * v;
  __syncthreads();
  for (int s = 128; s > 0; s >>= 1) {
    if (threadIdx.x < s) red[threadIdx.x] += red[threadIdx.x + s];
    __syncthreads();
  }
  if (threadIdx.x == 0) out[row] = red[0];
}

// ---------------------------------------------------------------------------
// Split fp32 -> (bf16 hi, bf16 lo) with RNE; hi+lo carries ~16 mantissa bits.
// ---------------------------------------------------------------------------
__device__ __forceinline__ unsigned short f2bf_rne(float f) {
  unsigned u = __float_as_uint(f);
  unsigned r = u + 0x7FFFu + ((u >> 16) & 1u);
  return (unsigned short)(r >> 16);
}

__global__ __launch_bounds__(256) void vq_split_bf16(
    const float* __restrict__ x, unsigned short* __restrict__ hi,
    unsigned short* __restrict__ lo, int n) {
  const int i = blockIdx.x * 256 + threadIdx.x;
  if (i >= n) return;
  const float f = x[i];
  const unsigned short h = f2bf_rne(f);
  const float fh = __uint_as_float((unsigned)h << 16);
  hi[i] = h;
  lo[i] = f2bf_rne(f - fh);
}

// ---------------------------------------------------------------------------
// Zero the persistent accumulators (avg_probs[K] + scalar sums).
// ---------------------------------------------------------------------------
__global__ void vq_init(float* __restrict__ avgp, float* __restrict__ scalars) {
  const int i = blockIdx.x * blockDim.x + threadIdx.x;
  if (i < K_) avgp[i] = 0.0f;
  if (i < 8) scalars[i] = 0.0f;
}

// ---------------------------------------------------------------------------
// Fragment loader for 16-bit 16x32 WMMA operands (A and B are symmetric):
// lane<16 -> K chunks {k0..k0+7} and {k0+16..k0+23}; lane>=16 -> +8 halves.
// Two contiguous 16-byte loads per fragment.
// ---------------------------------------------------------------------------
__device__ __forceinline__ v16bf load_frag(const __bf16* __restrict__ p) {
  v8bf c0 = *(const v8bf*)(p);
  v8bf c1 = *(const v8bf*)(p + 16);
  return __builtin_shufflevector(c0, c1, 0, 1, 2, 3, 4, 5, 6, 7,
                                 8, 9, 10, 11, 12, 13, 14, 15);
}

// ---------------------------------------------------------------------------
// Distance GEMM via compensated split-bf16 WMMA:
//   dot ~= zhi.ehi + zlo.ehi + zhi.elo   (fp32 accumulate in the matrix core)
//   d    = ||z||^2 + ||e||^2 - 2*dot
// Block = 256 threads = 8 waves; block tile = 64 rows x 128 cols.
// Wave tile = 16 rows x 64 cols = 4 fp32 accumulators of 16x16.
// ---------------------------------------------------------------------------
__global__ __launch_bounds__(256) void vq_dist_wmma(
    const unsigned short* __restrict__ zhi_u, const unsigned short* __restrict__ zlo_u,
    const unsigned short* __restrict__ ehi_u, const unsigned short* __restrict__ elo_u,
    const float* __restrict__ zsq, const float* __restrict__ esq,
    float* __restrict__ dchunk, int rowStart) {
  const __bf16* zhi = (const __bf16*)zhi_u;
  const __bf16* zlo = (const __bf16*)zlo_u;
  const __bf16* ehi = (const __bf16*)ehi_u;
  const __bf16* elo = (const __bf16*)elo_u;

  const int lane = threadIdx.x & 31;
  const int wave = threadIdx.x >> 5;
  const int half = lane >> 4;
  const int lr   = lane & 15;

  const int rowLocal = blockIdx.x * 64 + (wave & 3) * 16;   // row in chunk
  const int colBase  = blockIdx.y * 128 + (wave >> 2) * 64; // col in K_

  const size_t zoff = (size_t)(rowStart + rowLocal + lr) * D_ + 8 * half;
  const size_t eoff = (size_t)(colBase + lr) * D_ + 8 * half;

  v8f acc0 = {}; v8f acc1 = {}; v8f acc2 = {}; v8f acc3 = {};

#pragma unroll 2
  for (int k0 = 0; k0 < D_; k0 += 32) {
    v16bf ah = load_frag(zhi + zoff + k0);
    v16bf al = load_frag(zlo + zoff + k0);
    v16bf bh0 = load_frag(ehi + eoff + k0);
    v16bf bh1 = load_frag(ehi + eoff + 16 * D_ + k0);
    v16bf bh2 = load_frag(ehi + eoff + 32 * D_ + k0);
    v16bf bh3 = load_frag(ehi + eoff + 48 * D_ + k0);
    v16bf bl0 = load_frag(elo + eoff + k0);
    v16bf bl1 = load_frag(elo + eoff + 16 * D_ + k0);
    v16bf bl2 = load_frag(elo + eoff + 32 * D_ + k0);
    v16bf bl3 = load_frag(elo + eoff + 48 * D_ + k0);

    // hi*hi
    acc0 = __builtin_amdgcn_wmma_f32_16x16x32_bf16(false, ah, false, bh0, (short)0, acc0, false, false);
    acc1 = __builtin_amdgcn_wmma_f32_16x16x32_bf16(false, ah, false, bh1, (short)0, acc1, false, false);
    acc2 = __builtin_amdgcn_wmma_f32_16x16x32_bf16(false, ah, false, bh2, (short)0, acc2, false, false);
    acc3 = __builtin_amdgcn_wmma_f32_16x16x32_bf16(false, ah, false, bh3, (short)0, acc3, false, false);
    // lo*hi
    acc0 = __builtin_amdgcn_wmma_f32_16x16x32_bf16(false, al, false, bh0, (short)0, acc0, false, false);
    acc1 = __builtin_amdgcn_wmma_f32_16x16x32_bf16(false, al, false, bh1, (short)0, acc1, false, false);
    acc2 = __builtin_amdgcn_wmma_f32_16x16x32_bf16(false, al, false, bh2, (short)0, acc2, false, false);
    acc3 = __builtin_amdgcn_wmma_f32_16x16x32_bf16(false, al, false, bh3, (short)0, acc3, false, false);
    // hi*lo
    acc0 = __builtin_amdgcn_wmma_f32_16x16x32_bf16(false, ah, false, bl0, (short)0, acc0, false, false);
    acc1 = __builtin_amdgcn_wmma_f32_16x16x32_bf16(false, ah, false, bl1, (short)0, acc1, false, false);
    acc2 = __builtin_amdgcn_wmma_f32_16x16x32_bf16(false, ah, false, bl2, (short)0, acc2, false, false);
    acc3 = __builtin_amdgcn_wmma_f32_16x16x32_bf16(false, ah, false, bl3, (short)0, acc3, false, false);
  }

  // D layout: VGPR r -> row (r + 8*half), lane lr -> col (colTile + lr)
  const int outRow   = rowLocal + 8 * half;
  const int rowGBase = rowStart + outRow;
  const int c        = colBase + lr;
#pragma unroll
  for (int r = 0; r < 8; ++r) {
    const float zn = zsq[rowGBase + r];
    const size_t base = (size_t)(outRow + r) * K_;
    dchunk[base + c     ] = zn + esq[c     ] - 2.0f * acc0[r];
    dchunk[base + c + 16] = zn + esq[c + 16] - 2.0f * acc1[r];
    dchunk[base + c + 32] = zn + esq[c + 32] - 2.0f * acc2[r];
    dchunk[base + c + 48] = zn + esq[c + 48] - 2.0f * acc3[r];
  }
}

// ---------------------------------------------------------------------------
// Per-row reduction over K: argmin/min, logsumexp of flat=-d/T, and the
// sample-entropy term. One 256-thread block per row of the chunk.
// ---------------------------------------------------------------------------
__global__ __launch_bounds__(256) void vq_rowstats(
    const float* __restrict__ dchunk, float* __restrict__ rowMin,
    int* __restrict__ rowArg, float* __restrict__ rowLse,
    float* __restrict__ scalars, int rowStart) {
  __shared__ float svals[256];
  __shared__ int   sidx[256];
  const int rl = blockIdx.x;
  const float* drow = dchunk + (size_t)rl * K_;

  float bm = 3.4e38f;
  int   bi = 0;
  for (int k = threadIdx.x; k < K_; k += 256) {
    float v = drow[k];
    if (v < bm) { bm = v; bi = k; }
  }
  svals[threadIdx.x] = bm;
  sidx[threadIdx.x]  = bi;
  __syncthreads();
  for (int s = 128; s > 0; s >>= 1) {
    if (threadIdx.x < s) {
      float ov = svals[threadIdx.x + s]; int oi = sidx[threadIdx.x + s];
      float mv = svals[threadIdx.x];     int mi = sidx[threadIdx.x];
      if (ov < mv || (ov == mv && oi < mi)) {
        svals[threadIdx.x] = ov; sidx[threadIdx.x] = oi;
      }
    }
    __syncthreads();
  }
  const float dmin = svals[0];
  const int   amin = sidx[0];
  __syncthreads();

  // flat = -d/T ; m = -dmin/T ; S = sum exp(flat-m) ; T2 = sum flat*exp(flat-m)
  float ssum = 0.0f, tsum = 0.0f;
  for (int k = threadIdx.x; k < K_; k += 256) {
    float v = drow[k];
    float e = __expf((dmin - v) * INV_T);
    ssum += e;
    tsum += (-v * INV_T) * e;
  }
  svals[threadIdx.x] = ssum;
  __syncthreads();
  for (int s = 128; s > 0; s >>= 1) {
    if (threadIdx.x < s) svals[threadIdx.x] += svals[threadIdx.x + s];
    __syncthreads();
  }
  const float S = svals[0];
  __syncthreads();
  svals[threadIdx.x] = tsum;
  __syncthreads();
  for (int s = 128; s > 0; s >>= 1) {
    if (threadIdx.x < s) svals[threadIdx.x] += svals[threadIdx.x + s];
    __syncthreads();
  }
  const float T2 = svals[0];

  if (threadIdx.x == 0) {
    const int rg = rowStart + rl;
    rowMin[rg] = dmin;
    rowArg[rg] = amin;
    const float m   = -dmin * INV_T;
    const float lse = m + __logf(S);
    rowLse[rg] = lse;
    const float sampEnt = lse - T2 / S;     // -sum p*log p
    atomicAdd(&scalars[0], dmin);           // sum of min distances
    atomicAdd(&scalars[1], sampEnt);        // sum of per-row sample entropy
  }
}

// ---------------------------------------------------------------------------
// Column accumulation of avg_probs: avgp[k] += sum_rows exp(-d/T - lse_row).
// ---------------------------------------------------------------------------
__global__ __launch_bounds__(256) void vq_avgp(
    const float* __restrict__ dchunk, const float* __restrict__ rowLse,
    float* __restrict__ avgp, int rowStart, int nrows) {
  const int k = blockIdx.x * blockDim.x + threadIdx.x;
  float acc = 0.0f;
  for (int r = 0; r < nrows; ++r) {
    float d = dchunk[(size_t)r * K_ + k];
    acc += __expf(-d * INV_T - rowLse[rowStart + r]);
  }
  avgp[k] += acc;
}

// ---------------------------------------------------------------------------
// Final scalar losses.
// ---------------------------------------------------------------------------
__global__ __launch_bounds__(256) void vq_finalize(
    const float* __restrict__ avgp, const float* __restrict__ scalars,
    float* __restrict__ outScal) {
  __shared__ float red[256];
  float acc = 0.0f;
  for (int k = threadIdx.x; k < K_; k += 256) {
    float p = avgp[k] * (1.0f / (float)BN_);
    acc += p * __logf(p + 1e-5f);
  }
  red[threadIdx.x] = acc;
  __syncthreads();
  for (int s = 128; s > 0; s >>= 1) {
    if (threadIdx.x < s) red[threadIdx.x] += red[threadIdx.x + s];
    __syncthreads();
  }
  if (threadIdx.x == 0) {
    const float avg_entropy    = -red[0];
    const float sample_entropy = scalars[1] * (1.0f / (float)BN_);
    const float loss_commit    = scalars[0] * (1.0f / ((float)BN_ * (float)D_));
    const float loss_codebook  = loss_commit;   // same value, grads differ only
    const float loss_entropy   = sample_entropy - avg_entropy;
    const float loss = BETA_ * loss_commit + loss_codebook + ENTW_ * loss_entropy;
    outScal[0] = loss;
    outScal[1] = loss_commit;
    outScal[2] = loss_codebook;
    outScal[3] = loss_entropy;
    outScal[4] = sample_entropy;
    outScal[5] = avg_entropy;
  }
}

// ---------------------------------------------------------------------------
// Gather quantized vectors (straight-through output == quantized values) and
// emit indices as floats.
// ---------------------------------------------------------------------------
__global__ __launch_bounds__(256) void vq_gather(
    const float* __restrict__ emb, const int* __restrict__ rowArg,
    float* __restrict__ outQ, float* __restrict__ outIdx) {
  const size_t i = (size_t)blockIdx.x * blockDim.x + threadIdx.x;
  const size_t total = (size_t)BN_ * D_;
  if (i < total) {
    const int row = (int)(i / D_);
    const int c   = (int)(i % D_);
    outQ[i] = emb[(size_t)rowArg[row] * D_ + c];
  }
  if (i < BN_) outIdx[i] = (float)rowArg[i];
}

// ---------------------------------------------------------------------------
extern "C" void kernel_launch(void* const* d_in, const int* in_sizes, int n_in,
                              void* d_out, int out_size, void* d_ws, size_t ws_size,
                              hipStream_t stream) {
  const float* z   = (const float*)d_in[0];   // [B,N,D]
  const float* emb = (const float*)d_in[1];   // [K,D]

  float* out     = (float*)d_out;
  float* outQ    = out;                           // BN*D quantized_st
  float* outIdx  = out + (size_t)BN_ * D_;        // BN indices (as float)
  float* outScal = outIdx + BN_;                  // 6 scalars

  // workspace layout
  float* ws      = (float*)d_ws;
  float* zsq     = ws;                       // BN
  float* esq     = zsq + BN_;                // K
  float* rowMin  = esq + K_;                 // BN
  float* rowLse  = rowMin + BN_;             // BN
  int*   rowArg  = (int*)(rowLse + BN_);     // BN (ints)
  float* avgp    = (float*)(rowArg + BN_);   // K
  float* scalars = avgp + K_;                // 8
  unsigned short* zhi = (unsigned short*)(scalars + 8);  // BN*D bf16-hi
  unsigned short* zlo = zhi + (size_t)BN_ * D_;          // BN*D bf16-lo
  unsigned short* ehi = zlo + (size_t)BN_ * D_;          // K*D  bf16-hi
  unsigned short* elo = ehi + (size_t)K_ * D_;           // K*D  bf16-lo
  float* dchunk = (float*)(elo + (size_t)K_ * D_);       // chunkRows * K

  const size_t fixedBytes = (size_t)((char*)dchunk - (char*)d_ws);
  size_t availFloats = (ws_size > fixedBytes) ? (ws_size - fixedBytes) / sizeof(float) : 0;
  int chunkRows = (int)(availFloats / K_);
  if (chunkRows > BN_) chunkRows = BN_;
  chunkRows &= ~63;               // multiple of 64 for the 64-row block tile
  if (chunkRows < 64) chunkRows = 64;

  vq_rownorm<<<BN_, 256, 0, stream>>>(z, zsq);
  vq_rownorm<<<K_, 256, 0, stream>>>(emb, esq);
  vq_split_bf16<<<(BN_ * D_) / 256, 256, 0, stream>>>(z, zhi, zlo, BN_ * D_);
  vq_split_bf16<<<(K_ * D_) / 256, 256, 0, stream>>>(emb, ehi, elo, K_ * D_);
  vq_init<<<(K_ + 255) / 256, 256, 0, stream>>>(avgp, scalars);

  for (int rs = 0; rs < BN_; rs += chunkRows) {
    const int nr = (BN_ - rs < chunkRows) ? (BN_ - rs) : chunkRows;
    dim3 grid(nr / 64, K_ / 128);
    vq_dist_wmma<<<grid, 256, 0, stream>>>(zhi, zlo, ehi, elo, zsq, esq, dchunk, rs);
    vq_rowstats<<<nr, 256, 0, stream>>>(dchunk, rowMin, rowArg, rowLse, scalars, rs);
    vq_avgp<<<K_ / 256, 256, 0, stream>>>(dchunk, rowLse, avgp, rs, nr);
  }

  vq_finalize<<<1, 256, 0, stream>>>(avgp, scalars, outScal);

  const size_t total = (size_t)BN_ * D_;
  vq_gather<<<(unsigned)((total + 255) / 256), 256, 0, stream>>>(emb, rowArg, outQ, outIdx);
}
